// MoEFFN_86895778333203
// MI455X (gfx1250) — matrix operand order repaired and checked by reading, not compiled, not run on hardware
//
#include <hip/hip_runtime.h>
#include <hip/hip_bf16.h>
#include <math.h>

// Problem sizes (fixed by the reference)
#define NTOK 8192   // B*S = 4*2048
#define DDIM 512
#define FDIM 2048
#define NEXP 8
#define TM   64     // tokens per workgroup (64 => AI = 64 FLOP/byte of L2 weight traffic)
#define FC   128    // F-chunk processed per iteration
#define XP   520    // padded LDS pitch for X tile (bf16 elems): 1040B -> conflict-free b128 reads
#define HP   136    // padded LDS pitch for H tile (bf16 elems): 272B

typedef __attribute__((ext_vector_type(16))) __bf16 v16bf;
typedef __attribute__((ext_vector_type(8)))  float  v8f;

union FragBF {
    v16bf bf;
    uint4 q[2];
};

__device__ __forceinline__ unsigned short f2bf(float f) {
    unsigned int u = __float_as_uint(f);
    u += 0x7FFFu + ((u >> 16) & 1u);   // round-to-nearest-even
    return (unsigned short)(u >> 16);
}

__device__ __forceinline__ float gelu_tanh(float x) {
    float x3 = x * x * x;
    return 0.5f * x * (1.0f + tanhf(0.7978845608028654f * (x + 0.044715f * x3)));
}

// ---------------------------------------------------------------------------
// Prep: fp32 -> bf16 straight convert (for X)
// ---------------------------------------------------------------------------
__global__ void cvt_f32_bf16_kernel(const float* __restrict__ in,
                                    unsigned short* __restrict__ out, int n) {
    int i = blockIdx.x * blockDim.x + threadIdx.x;
    if (i < n) out[i] = f2bf(in[i]);
}

// ---------------------------------------------------------------------------
// Prep: batched tiled transpose fp32 [z][R][C] -> bf16 [z][C][R]
// ---------------------------------------------------------------------------
__global__ __launch_bounds__(256) void transpose_f32_bf16_kernel(
        const float* __restrict__ in, unsigned short* __restrict__ out,
        int R, int C) {
    __shared__ float tile[32][33];
    const float* src = in + (size_t)blockIdx.z * R * C;
    unsigned short* dst = out + (size_t)blockIdx.z * R * C;
    int x  = blockIdx.x * 32 + threadIdx.x;   // column index (C)
    int y0 = blockIdx.y * 32;                 // row base (R)
#pragma unroll
    for (int i = 0; i < 32; i += 8)
        tile[threadIdx.y + i][threadIdx.x] = src[(size_t)(y0 + threadIdx.y + i) * C + x];
    __syncthreads();
    int ox  = blockIdx.y * 32 + threadIdx.x;  // new column (= old row)
    int oy0 = blockIdx.x * 32;                // new row base (= old col)
#pragma unroll
    for (int i = 0; i < 32; i += 8)
        dst[(size_t)(oy0 + threadIdx.y + i) * R + ox] = f2bf(tile[threadIdx.x][threadIdx.y + i]);
}

// ---------------------------------------------------------------------------
// Fused expert FFN:  out[64tok, 512] (+)= gate * ( gelu(X@W1 + b1) @ W2 + b2 )
//   xbf : [NTOK, DDIM] bf16 row-major
//   w1t : [FDIM, DDIM] bf16 (transposed W1, row = output feature f)
//   w2t : [DDIM, FDIM] bf16 (transposed W2, row = output dim d)
//   gate: per-token gating prob with stride gstride (nullptr => 1.0)
// 512 threads = 16 waves (wave32). Wave w: column group (w&7), token-row half (w>>3).
// Paired waves (w, w+8) share identical weight B-fragments -> WGP$ hits, halving
// L2-side weight traffic per token vs a 32-token tile. Per-wave regs unchanged.
// Double-buffered H tile in LDS => one barrier per F-chunk.
// ---------------------------------------------------------------------------
__global__ __launch_bounds__(512) void moe_fused_ffn_kernel(
        const unsigned short* __restrict__ xbf,
        const unsigned short* __restrict__ w1t,
        const float* __restrict__ b1,
        const unsigned short* __restrict__ w2t,
        const float* __restrict__ b2,
        const float* __restrict__ gate, int gstride,
        float* __restrict__ outacc, int accumulate) {
    __shared__ unsigned short Xs[TM * XP];      // 64 x 512 bf16, padded   (66.5 KB)
    __shared__ unsigned short Hs[2][TM * HP];   // double-buffered 64x128  (34.8 KB)

    const int tid  = threadIdx.x;
    const int lane = tid & 31;
    const int wv   = tid >> 5;      // wave id 0..15
    const int cg   = wv & 7;        // column group 0..7
    const int rt   = wv >> 3;       // token-row half 0..1 (32 rows each)
    const int half = lane >> 4;     // 0/1 half-wave
    const int ln   = lane & 15;     // 0..15
    const int tokbase = blockIdx.x * TM;
    const int rowoff  = rt * 32;

    // ---- stage X tile into LDS (64 rows x 512 bf16) ----
    {
        const uint4* src = reinterpret_cast<const uint4*>(xbf + (size_t)tokbase * DDIM);
#pragma unroll
        for (int i = 0; i < 8; ++i) {
            int idx = tid + 512 * i;        // 4096 uint4 total
            int row = idx >> 6;             // /64 uint4 per row
            int c8  = idx & 63;
            uint4 v = src[row * (DDIM / 8) + c8];
            *reinterpret_cast<uint4*>(&Xs[row * XP + c8 * 8]) = v;
        }
    }
    __syncthreads();

    // output accumulators: [row-tile r][col-tile c]
    // wave owns rows [rowoff, rowoff+32), cols [cg*64, cg*64+64)
    v8f acc[2][4];
#pragma unroll
    for (int r = 0; r < 2; ++r)
#pragma unroll
        for (int c = 0; c < 4; ++c)
#pragma unroll
            for (int k = 0; k < 8; ++k) acc[r][c][k] = 0.0f;

#pragma unroll 1
    for (int fchunk = 0; fchunk < FDIM; fchunk += FC) {
        const int buf = (fchunk / FC) & 1;
        const int frow = fchunk + cg * 16 + ln;          // global output feature
        const unsigned short* w1row = w1t + (size_t)frow * DDIM;

        // prefetch next chunk's weights into cache while we compute this one
        if (fchunk + FC < FDIM) {
            __builtin_prefetch(w1t + (size_t)(frow + FC) * DDIM, 0, 3);
            __builtin_prefetch(w2t + (size_t)(cg * 64 + (lane >> 3) * 16 + ln) * FDIM
                                   + fchunk + FC + (lane & 8) * 8, 0, 3);
        }

        // ================= GEMM1: H = X @ W1chunk =================
        // wave computes H rows [rowoff, rowoff+32), cols [cg*16, cg*16+16)
        v8f h[2];
#pragma unroll
        for (int r = 0; r < 2; ++r)
#pragma unroll
            for (int k = 0; k < 8; ++k) h[r][k] = 0.0f;

        // ping-pong pipelined B fragments over the K=512 loop (16 k-steps)
        FragBF bA, bB;
        {
            const uint4* p = reinterpret_cast<const uint4*>(w1row + 0 + half * 16);
            bA.q[0] = p[0]; bA.q[1] = p[1];
        }
#pragma unroll 4
        for (int ks = 0; ks < DDIM / 32; ks += 2) {
            {
                const uint4* p = reinterpret_cast<const uint4*>(w1row + (ks + 1) * 32 + half * 16);
                bB.q[0] = p[0]; bB.q[1] = p[1];
            }
#pragma unroll
            for (int r = 0; r < 2; ++r) {
                FragBF afr;
                const unsigned short* xs =
                    &Xs[(rowoff + r * 16 + ln) * XP + ks * 32 + half * 16];
                afr.q[0] = *reinterpret_cast<const uint4*>(xs);
                afr.q[1] = *reinterpret_cast<const uint4*>(xs + 8);
                h[r] = __builtin_amdgcn_wmma_f32_16x16x32_bf16(
                    false, afr.bf, false, bA.bf, (short)0, h[r], false, false);
            }
            if (ks + 2 < DDIM / 32) {
                const uint4* p = reinterpret_cast<const uint4*>(w1row + (ks + 2) * 32 + half * 16);
                bA.q[0] = p[0]; bA.q[1] = p[1];
            }
#pragma unroll
            for (int r = 0; r < 2; ++r) {
                FragBF afr;
                const unsigned short* xs =
                    &Xs[(rowoff + r * 16 + ln) * XP + (ks + 1) * 32 + half * 16];
                afr.q[0] = *reinterpret_cast<const uint4*>(xs);
                afr.q[1] = *reinterpret_cast<const uint4*>(xs + 8);
                h[r] = __builtin_amdgcn_wmma_f32_16x16x32_bf16(
                    false, afr.bf, false, bB.bf, (short)0, h[r], false, false);
            }
        }

        // gelu + bias, write bf16 H tile to LDS buffer `buf`
        // (safe without a pre-barrier: chunk i-2 reads of this buffer are
        //  ordered before barrier(i-1), which every wave has passed)
        {
            const float bias = b1[frow];
#pragma unroll
            for (int r = 0; r < 2; ++r) {
#pragma unroll
                for (int v = 0; v < 8; ++v) {
                    int m = rowoff + r * 16 + half * 8 + v;  // token row within tile
                    Hs[buf][m * HP + cg * 16 + ln] = f2bf(gelu_tanh(h[r][v] + bias));
                }
            }
        }
        __syncthreads();

        // ================= GEMM2: acc += H @ W2chunk =================
        // ping-pong pipelined B fragments (4 per k-step) over K=128 (4 k-steps)
        FragBF b0[4], b1f[4];
#pragma unroll
        for (int c = 0; c < 4; ++c) {
            const int drow = cg * 64 + c * 16 + ln;
            const uint4* p = reinterpret_cast<const uint4*>(
                w2t + (size_t)drow * FDIM + fchunk + 0 + half * 16);
            b0[c].q[0] = p[0]; b0[c].q[1] = p[1];
        }
#pragma unroll
        for (int ks = 0; ks < FC / 32; ks += 2) {
#pragma unroll
            for (int c = 0; c < 4; ++c) {
                const int drow = cg * 64 + c * 16 + ln;
                const uint4* p = reinterpret_cast<const uint4*>(
                    w2t + (size_t)drow * FDIM + fchunk + (ks + 1) * 32 + half * 16);
                b1f[c].q[0] = p[0]; b1f[c].q[1] = p[1];
            }
            FragBF afr[2];
#pragma unroll
            for (int r = 0; r < 2; ++r) {
                const unsigned short* hs =
                    &Hs[buf][(rowoff + r * 16 + ln) * HP + ks * 32 + half * 16];
                afr[r].q[0] = *reinterpret_cast<const uint4*>(hs);
                afr[r].q[1] = *reinterpret_cast<const uint4*>(hs + 8);
            }
#pragma unroll
            for (int c = 0; c < 4; ++c)
#pragma unroll
                for (int r = 0; r < 2; ++r)
                    acc[r][c] = __builtin_amdgcn_wmma_f32_16x16x32_bf16(
                        false, afr[r].bf, false, b0[c].bf, (short)0, acc[r][c], false, false);
            if (ks + 2 < FC / 32) {
#pragma unroll
                for (int c = 0; c < 4; ++c) {
                    const int drow = cg * 64 + c * 16 + ln;
                    const uint4* p = reinterpret_cast<const uint4*>(
                        w2t + (size_t)drow * FDIM + fchunk + (ks + 2) * 32 + half * 16);
                    b0[c].q[0] = p[0]; b0[c].q[1] = p[1];
                }
            }
#pragma unroll
            for (int r = 0; r < 2; ++r) {
                const unsigned short* hs =
                    &Hs[buf][(rowoff + r * 16 + ln) * HP + (ks + 1) * 32 + half * 16];
                afr[r].q[0] = *reinterpret_cast<const uint4*>(hs);
                afr[r].q[1] = *reinterpret_cast<const uint4*>(hs + 8);
            }
#pragma unroll
            for (int c = 0; c < 4; ++c)
#pragma unroll
                for (int r = 0; r < 2; ++r)
                    acc[r][c] = __builtin_amdgcn_wmma_f32_16x16x32_bf16(
                        false, afr[r].bf, false, b1f[c].bf, (short)0, acc[r][c], false, false);
        }
        // no trailing barrier: next chunk writes the other H buffer
    }

    // ================= epilogue: bias2, gate, accumulate =================
#pragma unroll
    for (int r = 0; r < 2; ++r) {
        float g[8];
#pragma unroll
        for (int v = 0; v < 8; ++v) {
            int token = tokbase + rowoff + r * 16 + half * 8 + v;
            g[v] = gate ? gate[(size_t)token * gstride] : 1.0f;
        }
#pragma unroll
        for (int c = 0; c < 4; ++c) {
            const int d = cg * 64 + c * 16 + ln;
            const float bb = b2[d];
#pragma unroll
            for (int v = 0; v < 8; ++v) {
                int token = tokbase + rowoff + r * 16 + half * 8 + v;
                float val = (acc[r][c][v] + bb) * g[v];
                float* o = outacc + (size_t)token * DDIM + d;
                if (accumulate) *o += val;
                else            *o  = val;
            }
        }
    }
}

// ---------------------------------------------------------------------------
// Final blend: out = (1-sg)*expert_acc + sg*shared_acc
// ---------------------------------------------------------------------------
__global__ void combine_kernel(const float* __restrict__ acc,
                               const float* __restrict__ shacc,
                               const float* __restrict__ sgp,
                               float* __restrict__ out, int total) {
    int i = blockIdx.x * blockDim.x + threadIdx.x;
    if (i < total) {
        float sg = sgp[i / DDIM];
        out[i] = (1.0f - sg) * acc[i] + sg * shacc[i];
    }
}

// ---------------------------------------------------------------------------
extern "C" void kernel_launch(void* const* d_in, const int* in_sizes, int n_in,
                              void* d_out, int out_size, void* d_ws, size_t ws_size,
                              hipStream_t stream) {
    const float* X   = (const float*)d_in[0];   // [B,S,D]
    const float* GP  = (const float*)d_in[1];   // [B,S,E]
    const float* SGP = (const float*)d_in[2];   // [B,S,1]
    const float* W1  = (const float*)d_in[3];   // [E,D,F]
    const float* B1  = (const float*)d_in[4];   // [E,F]
    const float* W2  = (const float*)d_in[5];   // [E,F,D]
    const float* B2  = (const float*)d_in[6];   // [E,D]
    const float* WS1 = (const float*)d_in[7];   // [D,F]
    const float* BS1 = (const float*)d_in[8];   // [F]
    const float* WS2 = (const float*)d_in[9];   // [F,D]
    const float* BS2 = (const float*)d_in[10];  // [D]
    float* out = (float*)d_out;

    // workspace carve-up (~76 MB)
    char* ws = (char*)d_ws;
    unsigned short* xbf  = (unsigned short*)ws; ws += (size_t)NTOK * DDIM * 2;
    unsigned short* w1t  = (unsigned short*)ws; ws += (size_t)NEXP * FDIM * DDIM * 2;
    unsigned short* w2t  = (unsigned short*)ws; ws += (size_t)NEXP * DDIM * FDIM * 2;
    unsigned short* ws1t = (unsigned short*)ws; ws += (size_t)FDIM * DDIM * 2;
    unsigned short* ws2t = (unsigned short*)ws; ws += (size_t)DDIM * FDIM * 2;
    float* accbuf        = (float*)ws;          ws += (size_t)NTOK * DDIM * 4;
    float* shbuf         = (float*)ws;

    // prep: convert + transpose to bf16 (one-time, bandwidth-trivial)
    cvt_f32_bf16_kernel<<<(NTOK * DDIM) / 256, 256, 0, stream>>>(X, xbf, NTOK * DDIM);
    transpose_f32_bf16_kernel<<<dim3(FDIM / 32, DDIM / 32, NEXP), dim3(32, 8), 0, stream>>>(
        W1, w1t, DDIM, FDIM);
    transpose_f32_bf16_kernel<<<dim3(DDIM / 32, FDIM / 32, NEXP), dim3(32, 8), 0, stream>>>(
        W2, w2t, FDIM, DDIM);
    transpose_f32_bf16_kernel<<<dim3(FDIM / 32, DDIM / 32, 1), dim3(32, 8), 0, stream>>>(
        WS1, ws1t, DDIM, FDIM);
    transpose_f32_bf16_kernel<<<dim3(DDIM / 32, FDIM / 32, 1), dim3(32, 8), 0, stream>>>(
        WS2, ws2t, FDIM, DDIM);

    // experts 0..7 accumulate gated outputs into accbuf
    for (int e = 0; e < NEXP; ++e) {
        moe_fused_ffn_kernel<<<NTOK / TM, 512, 0, stream>>>(
            xbf,
            w1t + (size_t)e * FDIM * DDIM, B1 + (size_t)e * FDIM,
            w2t + (size_t)e * DDIM * FDIM, B2 + (size_t)e * DDIM,
            GP + e, NEXP, accbuf, e != 0);
    }
    // shared expert -> shbuf (gate = 1)
    moe_fused_ffn_kernel<<<NTOK / TM, 512, 0, stream>>>(
        xbf, ws1t, BS1, ws2t, BS2, nullptr, 1, shbuf, 0);

    // blend
    combine_kernel<<<(NTOK * DDIM) / 256, 256, 0, stream>>>(accbuf, shbuf, SGP, out,
                                                            NTOK * DDIM);
}
